// SymmetricRBM_56289841381589
// MI455X (gfx1250) — matrix operand order repaired and checked by reading, not compiled, not run on hardware
//
#include <hip/hip_runtime.h>
#include <stdint.h>

// MI455X / gfx1250 fused RBM contrastive-divergence kernel.
// All Gibbs state lives in LDS/registers; GEMMs use v_wmma_f32_16x16x32_f16.
// Both circulant weight matrices are pre-expanded into WMMA B-fragment layout
// in LDS once per block; every B operand is then two ds_load_b128.

typedef _Float16 v16h __attribute__((ext_vector_type(16)));
typedef _Float16 v8h  __attribute__((ext_vector_type(8)));
typedef float    v8f  __attribute__((ext_vector_type(8)));

#define NVIS 64
#define NALPHA 4
#define NHID 256            // NALPHA * NVIS
#define KSTEPS 10
#define ROWS_PER_WAVE 16
#define WAVES_PER_BLOCK 8
#define ROWS_PER_BLOCK 128  // WAVES_PER_BLOCK * ROWS_PER_WAVE
#define THREADS 256

// LDS pool layout (halves)
#define SV_OFF   0                       // [128*64]   visible state
#define SH_OFF   (SV_OFF + 8192)         // [8*16*32]  per-wave h staging
#define SWF_OFF  (SH_OFF + 4096)         // [32 frags * 512] forward weights
#define SWB_OFF  (SWF_OFF + 16384)       // [32 frags * 512] backward weights
#define SRED_OFF (SWB_OFF + 16384)       // float[256] reduction (reinterpreted)
#define SMEM_HALVES (SRED_OFF)
#define SMEM_BYTES  (SMEM_HALVES * 2 + THREADS * 4)

// ---------------- helpers ----------------

// Cheap stateless counter-based RNG: uniform in [0,1).
__device__ __forceinline__ float urand(uint32_t tag, uint32_t idx) {
  uint32_t h = idx * 0x9E3779B1u + tag * 0x85EBCA77u + 0x27220A95u;
  h ^= h >> 16; h *= 0x7FEB352Du;
  h ^= h >> 15; h *= 0x846CA68Bu;
  h ^= h >> 16;
  return (float)(h >> 8) * 0x1p-24f;
}

// Bernoulli(sigmoid(x)) without division:  u < 1/(1+e)  <=>  u*(1+e) < 1,
// where e = exp(-x).  e==+inf (x very negative) correctly rejects.
__device__ __forceinline__ _Float16 bern_sigmoid(float x, float u) {
  const float e = __expf(-x);
  return (_Float16)((__builtin_fmaf(u, e, u) < 1.0f) ? 1.0f : 0.0f);
}

// softplus(x) = max(x,0) + ln2 * log2(1 + exp(-|x|))  (hardware exp/log ops)
__device__ __forceinline__ float softplusf(float x) {
  const float e = __expf(-fabsf(x));
  return fmaxf(x, 0.0f) + 0.69314718056f * __log2f(1.0f + e);
}

// branchless 4-way select on uniform index (lowers to s_cselect chains)
__device__ __forceinline__ float sel4(float c0, float c1, float c2, float c3, int a) {
  const float lo = (a & 1) ? c1 : c0;
  const float hi = (a & 1) ? c3 : c2;
  return (a & 2) ? hi : lo;
}

__device__ __forceinline__ v8f wmma_f16(v16h a, v16h b, v8f c) {
  // D = A(16x32 f16) * B(32x16 f16) + C(16x16 f32)
  return __builtin_amdgcn_wmma_f32_16x16x32_f16(false, a, false, b, (short)0, c,
                                                false, false);
}

// A-fragment (16x32 f16) from an LDS row-major tile.
// Lane L (<16): row=L,    K = kb..kb+7 (lo) and kb+16..kb+23 (hi), kb=0
// Lane L (>=16): row=L-16, kb=8
__device__ __forceinline__ v16h load_afrag(const _Float16* rowbase, int koff) {
  const int lane = (int)(threadIdx.x & 31u);
  const int kb = (lane < 16) ? 0 : 8;
  const v8h* p = reinterpret_cast<const v8h*>(rowbase + koff + kb);
  const v8h lo = p[0];
  const v8h hi = p[2];   // +16 halves
  v16h r;
#pragma unroll
  for (int j = 0; j < 8; ++j) { r[j] = lo[j]; r[j + 8] = hi[j]; }
  return r;
}

// B-fragment from a fragment-layout LDS table: base[(fragIdx*32 + lane)*16 + j]
__device__ __forceinline__ v16h load_bfrag(const _Float16* base, int fragIdx) {
  const int lane = (int)(threadIdx.x & 31u);
  const v8h* p = reinterpret_cast<const v8h*>(base + (size_t)((fragIdx * 32 + lane) << 4));
  const v8h lo = p[0];
  const v8h hi = p[1];
  v16h r;
#pragma unroll
  for (int j = 0; j < 8; ++j) { r[j] = lo[j]; r[j + 8] = hi[j]; }
  return r;
}

// Lane partial of free energy over this wave's 16 rows:
//   part = -b * (lane's share of sum v) - sum softplus(wv + c) over lane's C slots
__device__ float fe_lane_partial(const _Float16* sV, const _Float16* sWf,
                                 float c0, float c1, float c2, float c3, float bs) {
  const int tid = (int)threadIdx.x;
  const int lane = tid & 31;
  const int wave = tid >> 5;
  const int colN = lane & 15;
  const _Float16* base = sV + (size_t)wave * ROWS_PER_WAVE * NVIS;

  float part = 0.0f;
  float vs = 0.0f;
#pragma unroll 4
  for (int e = lane; e < ROWS_PER_WAVE * NVIS; e += 32) vs += (float)base[e];
  part -= bs * vs;

  const _Float16* vrow = base + (size_t)colN * NVIS;
  const v16h a0 = load_afrag(vrow, 0);
  const v16h a1 = load_afrag(vrow, 32);
#pragma unroll 4
  for (int ct = 0; ct < 16; ++ct) {
    v8f acc = {};
    acc = wmma_f16(a0, load_bfrag(sWf, ct * 2 + 0), acc);
    acc = wmma_f16(a1, load_bfrag(sWf, ct * 2 + 1), acc);
    const float cc = sel4(c0, c1, c2, c3, ct >> 2);  // a = ct/4, lane-uniform
#pragma unroll
    for (int r = 0; r < 8; ++r) part -= softplusf(acc[r] + cc);
  }
  return part;
}

// ---------------- main kernel ----------------

__global__ __launch_bounds__(THREADS)
void rbm_cd_fused_kernel(const float* __restrict__ v_data,
                         const float* __restrict__ kernel_g,
                         const float* __restrict__ b_g,
                         const float* __restrict__ c_g,
                         float* __restrict__ out) {
  extern __shared__ __align__(16) _Float16 smem[];
  _Float16* sV   = smem + SV_OFF;    // [128][64]
  _Float16* sHst = smem + SH_OFF;    // [8][16][32]
  _Float16* sWf  = smem + SWF_OFF;   // 32 fragments, fragment layout
  _Float16* sWb  = smem + SWB_OFF;   // 32 fragments, fragment layout
  float*    sRed = reinterpret_cast<float*>(smem + SRED_OFF);

  const int tid = (int)threadIdx.x;
  const int lane = tid & 31;
  const int wave = tid >> 5;
  const int colN = lane & 15;
  const int mrow = (lane < 16) ? 0 : 8;  // C/D layout: M = r + mrow
  const int grow0 = (int)blockIdx.x * ROWS_PER_BLOCK + wave * ROWS_PER_WAVE;

  const float bs = b_g[0];
  const float c0 = c_g[0], c1 = c_g[1], c2 = c_g[2], c3 = c_g[3];

  // --- LDS prep: forward circulant in WMMA B-fragment layout ---
  // Wf[k=m][n=a*64+i] = kernel[a][(m-i) mod 64]; frag fi=ct*2+kt, lane, j:
  //   n = ct*16 + (lane&15);  k = kt*32 + (lane<16?0:8) + j + (j>=8?8:0)
  for (int id = tid; id < 32 * 32; id += THREADS) {
    const int fi = id >> 5, ln = id & 31;
    const int ct = fi >> 1, kt = fi & 1;
    const int ai = ct * 16 + (ln & 15);
    const int a = ai >> 6, i = ai & 63;
    const int kb = (ln < 16) ? 0 : 8;
    _Float16* dst = sWf + (size_t)id * 16;
#pragma unroll
    for (int j = 0; j < 16; ++j) {
      const int m = kt * 32 + kb + j + ((j >= 8) ? 8 : 0);
      dst[j] = (_Float16)kernel_g[a * 64 + ((m - i) & 63)];
    }
  }

  // --- LDS prep: backward circulant in WMMA B-fragment layout ---
  // Wb[k=a*64+m][n=i] = kernel[a][(i-1-m) mod 64]; frag fi=ctp*4+g, lane, j:
  //   n = g*16 + (lane&15);  k = ctp*32 + (lane<16?0:8) + j + (j>=8?8:0)
  for (int id = tid; id < 32 * 32; id += THREADS) {
    const int fi = id >> 5, ln = id & 31;
    const int ctp = fi >> 2, g = fi & 3;
    const int i = g * 16 + (ln & 15);
    const int kb = (ln < 16) ? 0 : 8;
    _Float16* dst = sWb + (size_t)id * 16;
#pragma unroll
    for (int j = 0; j < 16; ++j) {
      const int k = ctp * 32 + kb + j + ((j >= 8) ? 8 : 0);
      const int a = k >> 6, m = k & 63;
      dst[j] = (_Float16)kernel_g[a * 64 + ((i - 1 - m) & 63)];
    }
  }

  // --- initial visible state (wave-local rows) ---
  for (int e = lane; e < ROWS_PER_WAVE * NVIS; e += 32) {
    const int r = e >> 6, cc = e & 63;
    sV[(size_t)(wave * 16 + r) * NVIS + cc] =
        (_Float16)v_data[(size_t)(grow0 + r) * NVIS + cc];
  }
  __syncthreads();

  const _Float16* vrow = sV + (size_t)(wave * 16 + colN) * NVIS;
  const _Float16* hrow = sHst + (size_t)(wave * 16 + colN) * 32;

  v16h av0 = load_afrag(vrow, 0);
  v16h av1 = load_afrag(vrow, 32);

  // ---------------- K Gibbs sweeps, fully LDS/register resident ----------------
#pragma unroll 1
  for (int step = 0; step < KSTEPS; ++step) {
    const uint32_t tagH = 2u * (uint32_t)step;
    const uint32_t tagV = tagH + 1u;
    v8f wh[4];
#pragma unroll
    for (int g = 0; g < 4; ++g) wh[g] = (v8f){};

#pragma unroll 1
    for (int ctp = 0; ctp < 8; ++ctp) {
      const float cc = sel4(c0, c1, c2, c3, ctp >> 1);  // a = (ctp*2+hh)/4
      // produce a 16x32 K-slab of h in the per-wave staging buffer
#pragma unroll
      for (int hh = 0; hh < 2; ++hh) {
        const int ct = ctp * 2 + hh;
        v8f acc = {};
        acc = wmma_f16(av0, load_bfrag(sWf, ct * 2 + 0), acc);
        acc = wmma_f16(av1, load_bfrag(sWf, ct * 2 + 1), acc);
        const int ai = ct * 16 + colN;
#pragma unroll
        for (int r = 0; r < 8; ++r) {
          const int M = r + mrow;
          const float u = urand(tagH, (uint32_t)(grow0 + M) * NHID + (uint32_t)ai);
          sHst[(size_t)(wave * 16 + M) * 32 + hh * 16 + colN] =
              bern_sigmoid(acc[r] + cc, u);
        }
      }
      // consume the slab as A of GEMM2, accumulating wh over K
      const v16h ah = load_afrag(hrow, 0);
#pragma unroll
      for (int g = 0; g < 4; ++g)
        wh[g] = wmma_f16(ah, load_bfrag(sWb, ctp * 4 + g), wh[g]);
    }

    // sample v_next and write back wave-local rows of sV
#pragma unroll
    for (int g = 0; g < 4; ++g) {
#pragma unroll
      for (int r = 0; r < 8; ++r) {
        const int M = r + mrow;
        const float u =
            urand(tagV, (uint32_t)(grow0 + M) * NVIS + (uint32_t)(g * 16 + colN));
        sV[(size_t)(wave * 16 + M) * NVIS + g * 16 + colN] =
            bern_sigmoid(wh[g][r] + bs, u);
      }
    }
    av0 = load_afrag(vrow, 0);
    av1 = load_afrag(vrow, 32);
  }

  // ---------------- free energies ----------------
  float accum = -fe_lane_partial(sV, sWf, c0, c1, c2, c3, bs);  // -fe_model

  // reload original data rows (wave-local; DS counters order the reuse)
  for (int e = lane; e < ROWS_PER_WAVE * NVIS; e += 32) {
    const int r = e >> 6, cc = e & 63;
    sV[(size_t)(wave * 16 + r) * NVIS + cc] =
        (_Float16)v_data[(size_t)(grow0 + r) * NVIS + cc];
  }
  accum += fe_lane_partial(sV, sWf, c0, c1, c2, c3, bs);        // +fe_data

  // ---------------- block reduce + global accumulate ----------------
  sRed[tid] = accum;
  __syncthreads();
#pragma unroll
  for (int s = THREADS / 2; s > 0; s >>= 1) {
    if (tid < s) sRed[tid] += sRed[tid + s];
    __syncthreads();
  }
  if (tid == 0) atomicAdd(out, sRed[0] * (1.0f / 65536.0f));
}

// ---------------- host launch ----------------

extern "C" void kernel_launch(void* const* d_in, const int* in_sizes, int n_in,
                              void* d_out, int out_size, void* d_ws, size_t ws_size,
                              hipStream_t stream) {
  (void)n_in; (void)d_ws; (void)ws_size;
  const float* v_data  = (const float*)d_in[0];  // [B, 64]
  const float* kernel  = (const float*)d_in[1];  // [1, 4, 64]
  const float* b_scal  = (const float*)d_in[2];  // [1]
  const float* c_vec   = (const float*)d_in[3];  // [1, 4]
  float* out = (float*)d_out;

  hipMemsetAsync(out, 0, (size_t)out_size * sizeof(float), stream);

  const int rows = in_sizes[0] / NVIS;           // 65536
  const int grid = rows / ROWS_PER_BLOCK;        // 512
  rbm_cd_fused_kernel<<<dim3(grid), dim3(THREADS), SMEM_BYTES, stream>>>(
      v_data, kernel, b_scal, c_vec, out);
}
